// ModuleCorefProp_24790551232591
// MI455X (gfx1250) — matrix operand order repaired and checked by reading, not compiled, not run on hardware
//
#include <hip/hip_runtime.h>
#include <hip/hip_bf16.h>

// Problem constants (from the reference)
#define BB 1
#define TT 2048
#define WW 15
#define NN 512
#define DD 1024
#define HH 150
#define HP 160   // H padded to multiple of 32 for WMMA K / 16 for N
#define EE 20
#define NBK 10
#define PW 5     // waves per pairs block (10 n-tiles / 2 per wave)

typedef __attribute__((ext_vector_type(16))) _Float16 v16h;
typedef __attribute__((ext_vector_type(8)))  _Float16 v8h;
typedef __attribute__((ext_vector_type(8)))  float    v8f;
typedef __attribute__((ext_vector_type(4)))  float    v4f;

// ---------------------------------------------------------------------------
// Fast wave-level WMMA GEMM, guard-free hot loop:
//   C[M,N] (+)= A_f32[M,K] * Bt_f16[N,K]^T (+ bias[col], bias pre-padded)
// Preconditions: M%16==0, K%32==0, N%(16*NT)==0; Bt is N-major (K contiguous).
// One wave computes a 16 x (16*NT) tile: A fragment is reused NT times.
// ---------------------------------------------------------------------------
template <int NT>
__global__ __launch_bounds__(32)
void gemm_wmma_fast(const float* __restrict__ A, const _Float16* __restrict__ Bt,
                    float* __restrict__ C, const float* __restrict__ bias,
                    int K, int lda, int ldc, int acc)
{
    const int lane = threadIdx.x & 31;
    const int row0 = blockIdx.x * 16;
    const int col0 = blockIdx.y * (16 * NT);
    const int r    = lane & 15;
    const int kbA  = (lane < 16) ? 0 : 16;  // A fragment K-base (contiguous, matches B)
    const int kbB  = (lane < 16) ? 0 : 16;  // B fragment K-base

    const float* __restrict__ arow = A + (size_t)(row0 + r) * lda;

    const _Float16* __restrict__ bp[NT];
#pragma unroll
    for (int j = 0; j < NT; ++j)
        bp[j] = Bt + (size_t)(col0 + j * 16 + r) * K + kbB;

    v8f c[NT];
#pragma unroll
    for (int j = 0; j < NT; ++j) c[j] = (v8f){};
    if (acc) {
#pragma unroll
        for (int j = 0; j < NT; ++j) {
            int col = col0 + j * 16 + r;
#pragma unroll
            for (int v = 0; v < 8; ++v) {
                int row = row0 + ((lane < 16) ? v : v + 8);
                c[j][v] = C[(size_t)row * ldc + col];
            }
        }
    }

    for (int k0 = 0; k0 < K; k0 += 32) {
        if (k0 + 32 < K) __builtin_prefetch(arow + k0 + 32, 0, 1);  // global_prefetch_b8
        // A fragment: two groups of 8 consecutive f32 -> 4x b128 loads + pack
        v4f a0 = *(const v4f*)(arow + k0 + kbA);
        v4f a1 = *(const v4f*)(arow + k0 + kbA + 4);
        v4f a2 = *(const v4f*)(arow + k0 + kbA + 8);
        v4f a3 = *(const v4f*)(arow + k0 + kbA + 12);
        v16h a;
#pragma unroll
        for (int t = 0; t < 4; ++t) {
            a[t]      = (_Float16)a0[t];
            a[t + 4]  = (_Float16)a1[t];
            a[t + 8]  = (_Float16)a2[t];
            a[t + 12] = (_Float16)a3[t];
        }
#pragma unroll
        for (int j = 0; j < NT; ++j) {
            v16h b = *(const v16h*)(bp[j] + k0);   // 32B contiguous -> 2x b128
            c[j] = __builtin_amdgcn_wmma_f32_16x16x32_f16(false, a, false, b, (short)0, c[j], false, false);
        }
    }

#pragma unroll
    for (int j = 0; j < NT; ++j) {
        int col = col0 + j * 16 + r;
        float bsum = (bias != nullptr) ? bias[col] : 0.f;
#pragma unroll
        for (int v = 0; v < 8; ++v) {
            int row = row0 + ((lane < 16) ? v : v + 8);
            C[(size_t)row * ldc + col] = c[j][v] + bsum;
        }
    }
}

// ---------------------------------------------------------------------------
// Fused pair scorer. Block = PW waves handles one (i, 16-wide j-tile).
//   h0[jj,k] = relu(L[i,k] + R[j,k] + S[i,k]*S[j,k] + EW[bucket(i,j),k])  (LDS f16)
//   h1 = relu(h0 @ W1 + b1);  score[i,j] = h1 @ Wo + bo + ss[i] + ss[j]; diag 0
// W1t is transposed f16 [HP x HP]; b1P/WoP zero-padded to HP (no guards).
// The [N,N,H] tensor never exists in memory.
// ---------------------------------------------------------------------------
__global__ __launch_bounds__(32 * PW)
void pairs_fused(const float* __restrict__ L, const float* __restrict__ R,
                 const float* __restrict__ S, const float* __restrict__ EW,
                 const _Float16* __restrict__ W1t, const float* __restrict__ b1P,
                 const float* __restrict__ WoP, const float* __restrict__ bo,
                 const int* __restrict__ begin, const int* __restrict__ end,
                 const float* __restrict__ ss, float* __restrict__ scores)
{
    __shared__ _Float16 h0[16 * HP];
    __shared__ float    sacc[16];
    __shared__ int      jbuck[16];

    const int i   = blockIdx.y;
    const int j0  = blockIdx.x * 16;
    const int tid = threadIdx.x;

    if (tid < 16) {
        int j = j0 + tid;
        int d = begin[i] - end[j];
        if (d < 0) d = 0;
        int bk;
        if (d <= 4) bk = d;
        else {
            bk = (31 - __clz(d)) + 3;       // floor(log2(d)) + 3
            if (bk > NBK - 1) bk = NBK - 1;
        }
        jbuck[tid] = bk;
        sacc[tid]  = 0.f;
    }
    __syncthreads();

    // Build h0 tile in LDS (16 x 160, zero-padded K)
    for (int e = tid; e < 16 * HP; e += 32 * PW) {
        int jj = e / HP, k = e % HP;
        float v = 0.f;
        if (k < HH) {
            int j = j0 + jj;
            float hv = L[i * HP + k] + R[j * HP + k] + S[i * HP + k] * S[j * HP + k]
                     + EW[jbuck[jj] * HP + k];
            v = hv > 0.f ? hv : 0.f;
        }
        h0[e] = (_Float16)v;
    }
    __syncthreads();

    const int wave = tid >> 5, lane = tid & 31;
    const int r   = lane & 15;
    const int kbA = (lane < 16) ? 0 : 16;
    const int kbB = (lane < 16) ? 0 : 16;

    float acc[8];
#pragma unroll
    for (int v = 0; v < 8; ++v) acc[v] = 0.f;

    for (int nt = wave; nt < HP / 16; nt += PW) {        // exactly 2 tiles per wave
        const int col = nt * 16 + r;
        const _Float16* __restrict__ bptr = W1t + (size_t)col * HP + kbB;
        v8f c = {};
#pragma unroll
        for (int k0 = 0; k0 < HP; k0 += 32) {
            v8h alo = *(const v8h*)(&h0[r * HP + k0 + kbA]);       // ds_load_b128
            v8h ahi = *(const v8h*)(&h0[r * HP + k0 + kbA + 8]);
            v16h a;
#pragma unroll
            for (int t = 0; t < 8; ++t) { a[t] = alo[t]; a[t + 8] = ahi[t]; }
            v16h b = *(const v16h*)(bptr + k0);                    // 32B contiguous
            c = __builtin_amdgcn_wmma_f32_16x16x32_f16(false, a, false, b, (short)0, c, false, false);
        }
        const float wo = WoP[col];   // zero-padded: no guards, no exec juggling
        const float bb = b1P[col];
#pragma unroll
        for (int v = 0; v < 8; ++v) {
            float val = c[v] + bb;
            acc[v] += (val > 0.f) ? val * wo : 0.f;
        }
    }

    // One butterfly per row at the end (8 independent chains interleave)
#pragma unroll
    for (int v = 0; v < 8; ++v) {
        float val = acc[v];
#pragma unroll
        for (int off = 1; off < 16; off <<= 1)
            val += __shfl_xor(val, off, 32);
        if (r == 0)
            atomicAdd(&sacc[(lane < 16) ? v : v + 8], val);
    }
    __syncthreads();

    if (tid < 16) {
        int j = j0 + tid;
        float s = sacc[tid] + bo[0] + ss[i] + ss[j];
        if (i == j) s = 0.f;
        scores[i * NN + j] = s;
    }
}

// ---------------------------------------------------------------------------
// Row softmax with lower-triangular mask (j <= i), in place.
// ---------------------------------------------------------------------------
__global__ __launch_bounds__(256)
void softmax_tril(float* __restrict__ scores)
{
    const int i = blockIdx.x, tid = threadIdx.x;
    __shared__ float red[256];
    float m = -1e30f;
    for (int j = tid; j < NN; j += 256) {
        float v = (j <= i) ? scores[i * NN + j] : -1e30f;
        m = fmaxf(m, v);
    }
    red[tid] = m; __syncthreads();
    for (int s = 128; s > 0; s >>= 1) { if (tid < s) red[tid] = fmaxf(red[tid], red[tid + s]); __syncthreads(); }
    m = red[0]; __syncthreads();
    float sum = 0.f;
    for (int j = tid; j < NN; j += 256) {
        float e = (j <= i) ? __expf(scores[i * NN + j] - m) : 0.f;
        scores[i * NN + j] = e;
        sum += e;
    }
    red[tid] = sum; __syncthreads();
    for (int s = 128; s > 0; s >>= 1) { if (tid < s) red[tid] += red[tid + s]; __syncthreads(); }
    const float inv = 1.f / red[0];
    __syncthreads();
    for (int j = tid; j < NN; j += 256) scores[i * NN + j] *= inv;
}

// Transpose + convert to f16 with zero padding:
//   dst[n*Kp + k] = src[(rowoff+k)*srcld + n]  for k<Ksrc && n<Nsrc, else 0
__global__ void cvt_tr_f16(const float* __restrict__ src, _Float16* __restrict__ dst,
                           int Ksrc, int Nsrc, int Kp, int Np, int srcld, int rowoff)
{
    int idx = blockIdx.x * blockDim.x + threadIdx.x;
    int total = Np * Kp;
    if (idx >= total) return;
    int n = idx / Kp, k = idx % Kp;
    _Float16 v = (_Float16)0.f;
    if (k < Ksrc && n < Nsrc) v = (_Float16)src[(size_t)(rowoff + k) * srcld + n];
    dst[idx] = v;
}

// EW[b,k] = Edist[b] @ Wd[:,k] + bd[k], padded to 16 x HP
__global__ void ew_kernel(const float* __restrict__ Edist, const float* __restrict__ Wd,
                          const float* __restrict__ bd, float* __restrict__ EW)
{
    int b = blockIdx.x, k = threadIdx.x;  // grid 16 x HP threads
    float v = 0.f;
    if (b < NBK && k < HH) {
        float acc = bd[k];
        for (int e = 0; e < EE; ++e) acc += Edist[b * EE + e] * Wd[e * HH + k];
        v = acc;
    }
    EW[b * HP + k] = v;
}

// Zero-pad 5 length-HH f32 vectors to length-HP (bl, br, bp, b1, Wo)
__global__ void pad5_kernel(const float* __restrict__ s0, const float* __restrict__ s1,
                            const float* __restrict__ s2, const float* __restrict__ s3,
                            const float* __restrict__ s4,
                            float* __restrict__ d0, float* __restrict__ d1,
                            float* __restrict__ d2, float* __restrict__ d3,
                            float* __restrict__ d4)
{
    int b = blockIdx.x, k = threadIdx.x;   // grid 5 x HP threads
    const float* s = (b == 0) ? s0 : (b == 1) ? s1 : (b == 2) ? s2 : (b == 3) ? s3 : s4;
    float*       d = (b == 0) ? d0 : (b == 1) ? d1 : (b == 2) ? d2 : (b == 3) ? d3 : d4;
    d[k] = (k < HH) ? s[k] : 0.f;
}

// ss[i] = update[i] @ Wpr + bpr
__global__ __launch_bounds__(256)
void ss_kernel(const float* __restrict__ upd, const float* __restrict__ Wpr,
               const float* __restrict__ bpr, float* __restrict__ ss)
{
    const int i = blockIdx.x, tid = threadIdx.x;
    __shared__ float red[256];
    float acc = 0.f;
    for (int k = tid; k < DD; k += 256) acc += upd[i * DD + k] * Wpr[k];
    red[tid] = acc; __syncthreads();
    for (int s = 128; s > 0; s >>= 1) { if (tid < s) red[tid] += red[tid + s]; __syncthreads(); }
    if (tid == 0) ss[i] = red[0] + bpr[0];
}

// upd = sigmoid(gtmp)*upd + (1-sigmoid(gtmp))*ctxt
__global__ void gate_kernel(float* __restrict__ upd, const float* __restrict__ ctxt,
                            const float* __restrict__ gtmp, int n)
{
    int idx = blockIdx.x * blockDim.x + threadIdx.x;
    if (idx >= n) return;
    float g = 1.f / (1.f + __expf(-gtmp[idx]));
    upd[idx] = g * upd[idx] + (1.f - g) * ctxt[idx];
}

__global__ void copy_f32(const float* __restrict__ s, float* __restrict__ d, size_t n)
{
    size_t i = (size_t)blockIdx.x * blockDim.x + threadIdx.x;
    size_t st = (size_t)gridDim.x * blockDim.x;
    for (; i < n; i += st) d[i] = s[i];
}

// overwrite rows of flat all_span_vecs at prune_indices with update (masked by lengths)
__global__ __launch_bounds__(256)
void scatter_rows(const long long* __restrict__ idx, const int* __restrict__ lengths,
                  const float* __restrict__ upd, float* __restrict__ out)
{
    int kk = blockIdx.x;
    if (kk >= lengths[0]) return;
    long long f = idx[kk];
    for (int d = threadIdx.x; d < DD; d += 256)
        out[(size_t)f * DD + d] = upd[(size_t)kk * DD + d];
}

// ---------------------------------------------------------------------------
static inline size_t align256(size_t x) { return (x + 255) & ~size_t(255); }

extern "C" void kernel_launch(void* const* d_in, const int* in_sizes, int n_in,
                              void* d_out, int out_size, void* d_ws, size_t ws_size,
                              hipStream_t stream)
{
    (void)in_sizes; (void)n_in; (void)out_size; (void)ws_size;
    // Inputs (setup_inputs dict order)
    const float*     all_span   = (const float*)d_in[0];
    const float*     span_vecs  = (const float*)d_in[1];
    const float*     span_sc    = (const float*)d_in[2];
    /* triangular_mask d_in[3] implied by j<=i */
    const int*       begin      = (const int*)d_in[4];
    const int*       end        = (const int*)d_in[5];
    const long long* prune_idx  = (const long long*)d_in[6];
    const int*       lengths    = (const int*)d_in[7];
    const float *Wl = (const float*)d_in[8],  *bl = (const float*)d_in[9];
    const float *Wr = (const float*)d_in[10], *br = (const float*)d_in[11];
    const float *Wp = (const float*)d_in[12], *bp = (const float*)d_in[13];
    const float *Edist = (const float*)d_in[14];
    const float *Wd = (const float*)d_in[15], *bd = (const float*)d_in[16];
    const float *W1 = (const float*)d_in[17], *b1 = (const float*)d_in[18];
    const float *Wo = (const float*)d_in[19], *bo = (const float*)d_in[20];
    const float *Wg = (const float*)d_in[21], *bg = (const float*)d_in[22];
    const float *Wpr = (const float*)d_in[23], *bpr = (const float*)d_in[24];

    // Workspace carve-up
    char* p = (char*)d_ws;
    float* upd    = (float*)p;            p += align256(sizeof(float) * NN * DD);
    float* ctxt   = (float*)p;            p += align256(sizeof(float) * NN * DD);
    float* gtmp   = (float*)p;            p += align256(sizeof(float) * NN * DD);
    float* Lbuf   = (float*)p;            p += align256(sizeof(float) * NN * HP);
    float* Rbuf   = (float*)p;            p += align256(sizeof(float) * NN * HP);
    float* Sbuf   = (float*)p;            p += align256(sizeof(float) * NN * HP);
    float* EW     = (float*)p;            p += align256(sizeof(float) * 16 * HP);
    float* scores = (float*)p;            p += align256(sizeof(float) * NN * NN);
    float* ssb    = (float*)p;            p += align256(sizeof(float) * NN);
    float* blP    = (float*)p;            p += align256(sizeof(float) * HP);
    float* brP    = (float*)p;            p += align256(sizeof(float) * HP);
    float* bpP    = (float*)p;            p += align256(sizeof(float) * HP);
    float* b1P    = (float*)p;            p += align256(sizeof(float) * HP);
    float* WoP    = (float*)p;            p += align256(sizeof(float) * HP);
    _Float16* updt = (_Float16*)p;        p += align256(sizeof(_Float16) * DD * NN);      // [D, N] transposed
    _Float16* Wlt  = (_Float16*)p;        p += align256(sizeof(_Float16) * HP * DD);      // [HP, D]
    _Float16* Wrt  = (_Float16*)p;        p += align256(sizeof(_Float16) * HP * DD);
    _Float16* Wpt  = (_Float16*)p;        p += align256(sizeof(_Float16) * HP * DD);
    _Float16* W1t  = (_Float16*)p;        p += align256(sizeof(_Float16) * HP * HP);      // [HP, HP]
    _Float16* Wgt0 = (_Float16*)p;        p += align256(sizeof(_Float16) * DD * DD);      // rows 0..D-1 of Wg, transposed
    _Float16* Wgt1 = (_Float16*)p;        p += align256(sizeof(_Float16) * DD * DD);      // rows D..2D-1

    // --- weight prep: transposed, zero-padded f16 copies (K-contiguous) ---
    {
        int t1 = HP * DD;   // projection weights: Ksrc=D, Nsrc=H -> [HP, D]
        cvt_tr_f16<<<(t1 + 255) / 256, 256, 0, stream>>>(Wl, Wlt, DD, HH, DD, HP, HH, 0);
        cvt_tr_f16<<<(t1 + 255) / 256, 256, 0, stream>>>(Wr, Wrt, DD, HH, DD, HP, HH, 0);
        cvt_tr_f16<<<(t1 + 255) / 256, 256, 0, stream>>>(Wp, Wpt, DD, HH, DD, HP, HH, 0);
        int t2 = HP * HP;
        cvt_tr_f16<<<(t2 + 255) / 256, 256, 0, stream>>>(W1, W1t, HH, HH, HP, HP, HH, 0);
        int t3 = DD * DD;
        cvt_tr_f16<<<(t3 + 255) / 256, 256, 0, stream>>>(Wg, Wgt0, DD, DD, DD, DD, DD, 0);
        cvt_tr_f16<<<(t3 + 255) / 256, 256, 0, stream>>>(Wg, Wgt1, DD, DD, DD, DD, DD, DD);
        ew_kernel<<<16, HP, 0, stream>>>(Edist, Wd, bd, EW);
        pad5_kernel<<<5, HP, 0, stream>>>(bl, br, bp, b1, Wo, blP, brP, bpP, b1P, WoP);
        copy_f32<<<2048, 256, 0, stream>>>(span_vecs, upd, (size_t)NN * DD);
    }

    dim3 gProj(NN / 16, HP / 32);    // NT=2 -> 32 cols per wave
    dim3 gBig (NN / 16, DD / 64);    // NT=4 -> 64 cols per wave
    dim3 gPairs(NN / 16, NN);

    // --- initial projections + pair scores (ss = span_scores) ---
    gemm_wmma_fast<2><<<gProj, 32, 0, stream>>>(upd, Wlt, Lbuf, blP, DD, DD, HP, 0);
    gemm_wmma_fast<2><<<gProj, 32, 0, stream>>>(upd, Wrt, Rbuf, brP, DD, DD, HP, 0);
    gemm_wmma_fast<2><<<gProj, 32, 0, stream>>>(upd, Wpt, Sbuf, bpP, DD, DD, HP, 0);
    pairs_fused<<<gPairs, 32 * PW, 0, stream>>>(Lbuf, Rbuf, Sbuf, EW, W1t, b1P, WoP, bo,
                                                begin, end, span_sc, scores);

    // --- CP = 2 coref-prop iterations ---
    for (int it = 0; it < 2; ++it) {
        softmax_tril<<<NN, 256, 0, stream>>>(scores);
        // updt[d, j] = upd[j, d]  (B operand for ctxt GEMM, K=NN contiguous)
        cvt_tr_f16<<<(DD * NN + 255) / 256, 256, 0, stream>>>(upd, updt, NN, DD, NN, DD, DD, 0);
        // ctxt = probs @ update : M=NN, K=NN, N=DD
        gemm_wmma_fast<4><<<gBig, 32, 0, stream>>>(scores, updt, ctxt, nullptr, NN, NN, DD, 0);
        // gate pre-activation: gtmp = upd@Wg[:D] + bg ; gtmp += ctxt@Wg[D:]
        gemm_wmma_fast<4><<<gBig, 32, 0, stream>>>(upd,  Wgt0, gtmp, bg,      DD, DD, DD, 0);
        gemm_wmma_fast<4><<<gBig, 32, 0, stream>>>(ctxt, Wgt1, gtmp, nullptr, DD, DD, DD, 1);
        gate_kernel<<<(NN * DD + 255) / 256, 256, 0, stream>>>(upd, ctxt, gtmp, NN * DD);

        // re-project + re-score with ss = upd @ Wpr + bpr
        gemm_wmma_fast<2><<<gProj, 32, 0, stream>>>(upd, Wlt, Lbuf, blP, DD, DD, HP, 0);
        gemm_wmma_fast<2><<<gProj, 32, 0, stream>>>(upd, Wrt, Rbuf, brP, DD, DD, HP, 0);
        gemm_wmma_fast<2><<<gProj, 32, 0, stream>>>(upd, Wpt, Sbuf, bpP, DD, DD, HP, 0);
        ss_kernel<<<NN, 256, 0, stream>>>(upd, Wpr, bpr, ssb);
        pairs_fused<<<gPairs, 32 * PW, 0, stream>>>(Lbuf, Rbuf, Sbuf, EW, W1t, b1P, WoP, bo,
                                                    begin, end, ssb, scores);
    }

    // --- assemble outputs: [all_out | update | scores] ---
    float* out = (float*)d_out;
    const size_t n_all = (size_t)BB * TT * WW * DD;      // 31457280
    copy_f32<<<8192, 256, 0, stream>>>(all_span, out, n_all);
    scatter_rows<<<NN, 256, 0, stream>>>(prune_idx, lengths, upd, out);
    copy_f32<<<2048, 256, 0, stream>>>(upd, out + n_all, (size_t)NN * DD);
    copy_f32<<<1024, 256, 0, stream>>>(scores, out + n_all + (size_t)NN * DD, (size_t)NN * NN);
}
